// MoEINR_14104672600553
// MI455X (gfx1250) — matrix operand order repaired and checked by reading, not compiled, not run on hardware
//
#include <hip/hip_runtime.h>
#include <hip/hip_bf16.h>

// ---------------------------------------------------------------------------
// MoE-INR fully-fused inference for MI455X (gfx1250, wave32, WMMA).
//
// Roofline: ~138 GFLOP vs ~5 MB traffic -> purely compute bound. Strategy:
//  * f16 weights/activations, f32 WMMA accumulation (v_wmma_f32_16x16x32_f16)
//  * whole network fused in one kernel; activations staged in LDS
//    (~303 KB of the 320 KB WGP LDS), weights served from the 192 MB L2
//  * 8 waves/block, each wave owns a 16-row M tile (128 rows per block)
//  * A-fragments hoisted to registers once per layer; two independent
//    N-tile accumulator chains per iteration to cover WMMA D->C latency
// ---------------------------------------------------------------------------

typedef __attribute__((ext_vector_type(16))) _Float16 v16h;
typedef __attribute__((ext_vector_type(8)))  _Float16 v8h;
typedef __attribute__((ext_vector_type(8)))  float    v8f;

#define OMEGA_F 30.0f
#define PI_F    3.14159265358979323846f

constexpr int BROWS = 128;   // rows per block (8 waves x 16)
constexpr int ST    = 264;   // f16 element stride of 256-wide LDS buffers (+8 pad)
constexpr int PFST  = 136;   // f16 element stride of 128-wide pf buffer (+8 pad)

// ---- f16 weight arena layout in d_ws (element offsets) --------------------
constexpr size_t OFF_ENC1 = 0;                         // [128][64]
constexpr size_t OFF_ENC2 = OFF_ENC1 + 128 * 64;       // [256][128]
constexpr size_t OFF_FC1  = OFF_ENC2 + 256 * 128;      // [128][256]
constexpr size_t OFF_FC2  = OFF_FC1  + 128 * 256;      // [128][128]
constexpr size_t OFF_FC3  = OFF_FC2  + 128 * 128;      // [256][128]
constexpr size_t OFF_POL1 = OFF_FC3  + 256 * 128;      // [128][32] (4 valid cols)
constexpr size_t OFF_POL2 = OFF_POL1 + 128 * 32;       // [128][128]
constexpr size_t OFF_GATE = OFF_POL2 + 128 * 128;      // [16][384] (7 valid rows)
constexpr size_t OFF_E1   = OFF_GATE + 16 * 384;       // 7 x [256][256]
constexpr size_t OFF_E2   = OFF_E1   + 7 * 256 * 256;  // 7 x [256][256]
constexpr size_t OFF_FIN  = OFF_E2   + 7 * 256 * 256;  // 7 x [16][256] (1 valid row)
constexpr size_t W_TOTAL  = OFF_FIN  + 7 * 16 * 256;   // 1,095,680 f16 = 2.09 MB

// ---- LDS layout (bytes) ---------------------------------------------------
constexpr size_t SM_A     = 0;                               // ping
constexpr size_t SM_B     = SM_A + BROWS * ST * 2;           // pong
constexpr size_t SM_H     = SM_B + BROWS * ST * 2;           // encoder h (residual src)
constexpr size_t SM_E     = SM_H + BROWS * ST * 2;           // enc_feat (expert input)
constexpr size_t SM_PF    = SM_E + BROWS * ST * 2;           // policy features [128]
constexpr size_t SM_PROBS = SM_PF + BROWS * PFST * 2;        // [128][8] f32
constexpr size_t SM_Y     = SM_PROBS + BROWS * 8 * 4;        // [128] f32
constexpr size_t SM_TOTAL = SM_Y + BROWS * 4;                // = 309,760 B

// ---------------------------------------------------------------------------
// fp32 -> f16 weight conversion with zero padding
// ---------------------------------------------------------------------------
__global__ void cvt_pad_f16(const float* __restrict__ s, _Float16* __restrict__ d,
                            int rs, int cs, int rd, int cd) {
  int i = blockIdx.x * blockDim.x + threadIdx.x;
  int n = rd * cd;
  if (i >= n) return;
  int r = i / cd, c = i % cd;
  float v = (r < rs && c < cs) ? s[(size_t)r * cs + c] : 0.0f;
  d[i] = (_Float16)v;
}

// ---------------------------------------------------------------------------
// WMMA fragment loaders (ISA 7.12.2 layouts, wave32)
// A 16x32 f16: lane m=lane&15 row; half=lane>>4 selects K chunks
//   VGPR0-3 <- K = kBase + 8*half .. +7 ; VGPR4-7 <- K = kBase+16+8*half .. +7
// B 32x16 f16 (B[k][n] = W[n][k]): lane n=lane&15 column; same K order as A:
//   elems 0-7 <- K = kBase + 8*half .. +7 ; elems 8-15 <- K = kBase+16+8*half .. +7
// ---------------------------------------------------------------------------
__device__ __forceinline__ v16h load_a_frag(const _Float16* act, int stride,
                                            int rBase, int kBase, int lane) {
  const int m = lane & 15, half = lane >> 4;
  const _Float16* row = act + (size_t)(rBase + m) * stride;
  v8h c0 = *(const v8h*)(row + kBase + 8 * half);
  v8h c1 = *(const v8h*)(row + kBase + 16 + 8 * half);
  return __builtin_shufflevector(c0, c1, 0, 1, 2, 3, 4, 5, 6, 7,
                                         8, 9, 10, 11, 12, 13, 14, 15);
}

__device__ __forceinline__ v16h load_b_frag(const _Float16* W, int Kdim,
                                            int nBase, int kBase, int lane) {
  const int n = nBase + (lane & 15);
  const _Float16* p = W + (size_t)n * Kdim + kBase + 8 * (lane >> 4);
  v8h lo = *(const v8h*)p;
  v8h hi = *(const v8h*)(p + 16);
  return __builtin_shufflevector(lo, hi, 0, 1, 2, 3, 4, 5, 6, 7,
                                         8, 9, 10, 11, 12, 13, 14, 15);
}

// ---------------------------------------------------------------------------
// Epilogue for one 16x16 C tile: bias [+res], activation, f16 store to LDS.
// ---------------------------------------------------------------------------
template <int ACT, bool RES>
__device__ __forceinline__ void store_tile(v8f c, float bv,
                                           _Float16* __restrict__ dst, int dStride,
                                           const _Float16* __restrict__ res, int resStride,
                                           int rBase, int n, int half) {
#pragma unroll
  for (int r = 0; r < 8; ++r) {
    const int row = rBase + r + 8 * half;   // C layout: VGPR r -> M = r + 8*half
    float v = c[r] + bv;
    if (RES) v += (float)res[(size_t)row * resStride + n];
    v = (ACT == 0) ? __sinf(OMEGA_F * v) : fmaxf(v, 0.0f);
    dst[(size_t)row * dStride + n] = (_Float16)v;
  }
}

// ---------------------------------------------------------------------------
// One dense layer for this wave's 16-row tile: dst = act( src @ W^T + b [+res] )
// ACT: 0 = sin(30x), 1 = relu. A-fragments hoisted; two N tiles in flight.
// Ndim must be a multiple of 32.
// ---------------------------------------------------------------------------
template <int KD, int ACT, bool RES>
__device__ __forceinline__ void mlp_layer(const _Float16* __restrict__ src, int sStride,
                                          const _Float16* __restrict__ W,
                                          const float* __restrict__ bias,
                                          int Ndim,
                                          _Float16* __restrict__ dst, int dStride,
                                          const _Float16* __restrict__ res, int resStride,
                                          int rBase, int lane) {
  constexpr int NK = KD / 32;
  v16h af[NK];
#pragma unroll
  for (int i = 0; i < NK; ++i) af[i] = load_a_frag(src, sStride, rBase, i * 32, lane);

  const int half = lane >> 4;
  const int n0   = lane & 15;
  for (int nb = 0; nb < Ndim; nb += 32) {
    v8f c0 = {}, c1 = {};
#pragma unroll
    for (int i = 0; i < NK; ++i) {
      v16h b0 = load_b_frag(W, KD, nb,      i * 32, lane);
      v16h b1 = load_b_frag(W, KD, nb + 16, i * 32, lane);
      c0 = __builtin_amdgcn_wmma_f32_16x16x32_f16(false, af[i], false, b0,
                                                  (short)0, c0, false, false);
      c1 = __builtin_amdgcn_wmma_f32_16x16x32_f16(false, af[i], false, b1,
                                                  (short)0, c1, false, false);
    }
    store_tile<ACT, RES>(c0, bias[nb + n0],      dst, dStride, res, resStride,
                         rBase, nb + n0, half);
    store_tile<ACT, RES>(c1, bias[nb + 16 + n0], dst, dStride, res, resStride,
                         rBase, nb + 16 + n0, half);
  }
}

// ---------------------------------------------------------------------------
// Fused network kernel
// ---------------------------------------------------------------------------
struct MoEParams {
  const float* x;
  const float *b_enc1, *b_enc2, *b_fc1, *b_fc2, *b_fc3;
  const float *b_pol1, *b_pol2, *b_gate;
  const float *b_e1, *b_e2, *b_fin;   // [7][256],[7][256],[7][1]
  const _Float16* w;                  // f16 weight arena (d_ws)
  float* out;                         // [B][1]
};

__global__ __launch_bounds__(256) void moe_inr_fused(MoEParams p) {
  extern __shared__ char smem[];
  _Float16* A     = (_Float16*)(smem + SM_A);
  _Float16* Bb    = (_Float16*)(smem + SM_B);
  _Float16* H     = (_Float16*)(smem + SM_H);
  _Float16* E     = (_Float16*)(smem + SM_E);
  _Float16* PF    = (_Float16*)(smem + SM_PF);
  float*    probs = (float*)(smem + SM_PROBS);
  float*    yacc  = (float*)(smem + SM_Y);

  const int tid   = threadIdx.x;
  const int lane  = tid & 31;
  const int wave  = tid >> 5;
  const int rBase = wave * 16;        // this wave's local row tile
  const int half  = lane >> 4;
  const int m     = lane & 15;
  const int gBase = blockIdx.x * BROWS;

  if (tid < BROWS) yacc[tid] = 0.0f;

  // ---- positional encoding: x[4] -> pe[64] into A -------------------------
  {
    const int row = rBase + m;
    float x4[4];
#pragma unroll
    for (int d = 0; d < 4; ++d) x4[d] = p.x[(size_t)(gBase + row) * 4 + d];
#pragma unroll
    for (int j = 0; j < 32; ++j) {
      const int   idx = 32 * half + j;        // pe column 0..63
      const int   d   = idx >> 4;
      const int   t   = idx & 15;             // 0-7 sin, 8-15 cos
      const float ang = x4[d] * (PI_F * (float)(1 << (t & 7)));
      const float v   = (t < 8) ? __sinf(ang) : __cosf(ang);
      A[(size_t)row * ST + idx] = (_Float16)v;
    }
  }
  __syncthreads();

  // ---- shared encoder -----------------------------------------------------
  mlp_layer<64, 0, false>(A, ST, p.w + OFF_ENC1, p.b_enc1, 128, Bb, ST, nullptr, 0, rBase, lane);
  __syncthreads();
  mlp_layer<128, 0, false>(Bb, ST, p.w + OFF_ENC2, p.b_enc2, 256, H, ST, nullptr, 0, rBase, lane);
  __syncthreads();
  mlp_layer<256, 1, false>(H, ST, p.w + OFF_FC1, p.b_fc1, 128, A, ST, nullptr, 0, rBase, lane);
  __syncthreads();
  mlp_layer<128, 1, false>(A, ST, p.w + OFF_FC2, p.b_fc2, 128, Bb, ST, nullptr, 0, rBase, lane);
  __syncthreads();
  // fc3 + residual(h) + relu -> enc_feat
  mlp_layer<128, 1, true>(Bb, ST, p.w + OFF_FC3, p.b_fc3, 256, E, ST, H, ST, rBase, lane);
  __syncthreads();

  // ---- policy net: x (padded to K=32) -> 128 -> 128 -----------------------
  {
    const int row = rBase + m;
#pragma unroll
    for (int i = 0; i < 16; ++i) {
      const int   cc = 16 * half + i;
      const float v  = (cc < 4) ? p.x[(size_t)(gBase + row) * 4 + cc] : 0.0f;
      A[(size_t)row * ST + cc] = (_Float16)v;
    }
  }
  __syncthreads();
  mlp_layer<32, 0, false>(A, ST, p.w + OFF_POL1, p.b_pol1, 128, Bb, ST, nullptr, 0, rBase, lane);
  __syncthreads();
  mlp_layer<128, 0, false>(Bb, ST, p.w + OFF_POL2, p.b_pol2, 128, PF, PFST, nullptr, 0, rBase, lane);
  __syncthreads();

  // ---- gate: concat(enc_feat[256], pf[128]) @ gate^T (7 of 16 cols) -------
  {
    v16h aE[8], aP[4];
#pragma unroll
    for (int i = 0; i < 8; ++i) aE[i] = load_a_frag(E, ST, rBase, i * 32, lane);
#pragma unroll
    for (int i = 0; i < 4; ++i) aP[i] = load_a_frag(PF, PFST, rBase, i * 32, lane);
    v8f c = {};
#pragma unroll
    for (int i = 0; i < 8; ++i) {
      v16h b = load_b_frag(p.w + OFF_GATE, 384, 0, i * 32, lane);
      c = __builtin_amdgcn_wmma_f32_16x16x32_f16(false, aE[i], false, b, (short)0, c, false, false);
    }
#pragma unroll
    for (int i = 0; i < 4; ++i) {
      v16h b = load_b_frag(p.w + OFF_GATE, 384, 0, 256 + i * 32, lane);
      c = __builtin_amdgcn_wmma_f32_16x16x32_f16(false, aP[i], false, b, (short)0, c, false, false);
    }
    if (m < 7) {
      const float bv = p.b_gate[m];
#pragma unroll
      for (int r = 0; r < 8; ++r) {
        const int row = rBase + r + 8 * half;
        probs[row * 8 + m] = c[r] + bv;
      }
    }
  }
  __syncthreads();

  // ---- softmax over 7 logits, one row per lane 0..15 of the owning wave ---
  if (lane < 16) {
    float* pr = &probs[(rBase + lane) * 8];
    float  mx = pr[0];
#pragma unroll
    for (int i = 1; i < 7; ++i) mx = fmaxf(mx, pr[i]);
    float e7[7], s = 0.0f;
#pragma unroll
    for (int i = 0; i < 7; ++i) { e7[i] = __expf(pr[i] - mx); s += e7[i]; }
    const float inv = 1.0f / s;
#pragma unroll
    for (int i = 0; i < 7; ++i) pr[i] = e7[i] * inv;
  }
  __syncthreads();

  // ---- 7 expert SIRENs (bulk of the FLOPs), soft-routed accumulate --------
  for (int e = 0; e < 7; ++e) {
    // Warm the cache with the next expert's two big matrices (128 KB each).
    if (e + 1 < 7) {
      const char* nw = (const char*)(p.w + OFF_E1 + (size_t)(e + 1) * 256 * 256);
      __builtin_prefetch(nw + (size_t)tid * 512, 0, 0);
      const char* nw2 = (const char*)(p.w + OFF_E2 + (size_t)(e + 1) * 256 * 256);
      __builtin_prefetch(nw2 + (size_t)tid * 512, 0, 0);
    }
    mlp_layer<256, 0, false>(E, ST, p.w + OFF_E1 + (size_t)e * 256 * 256,
                             p.b_e1 + e * 256, 256, A, ST, nullptr, 0, rBase, lane);
    __syncthreads();
    mlp_layer<256, 0, false>(A, ST, p.w + OFF_E2 + (size_t)e * 256 * 256,
                             p.b_e2 + e * 256, 256, Bb, ST, nullptr, 0, rBase, lane);
    __syncthreads();
    // final 256 -> 1 (padded to an N=16 WMMA tile; column n==0 is valid)
    v16h af[8];
#pragma unroll
    for (int i = 0; i < 8; ++i) af[i] = load_a_frag(Bb, ST, rBase, i * 32, lane);
    v8f c = {};
#pragma unroll
    for (int i = 0; i < 8; ++i) {
      v16h b = load_b_frag(p.w + OFF_FIN + (size_t)e * 16 * 256, 256, 0, i * 32, lane);
      c = __builtin_amdgcn_wmma_f32_16x16x32_f16(false, af[i], false, b, (short)0, c, false, false);
    }
    if (m == 0) {                       // lanes 0 and 16 hold column N=0
      const float fb = p.b_fin[e];
#pragma unroll
      for (int r = 0; r < 8; ++r) {
        const int row = rBase + r + 8 * half;
        yacc[row] += probs[row * 8 + e] * (c[r] + fb);
      }
    }
    __syncthreads();
  }

  if (lane < 16) p.out[gBase + rBase + lane] = yacc[rBase + lane];
}

// ---------------------------------------------------------------------------
// Host launch: weight-conversion prepass + fused kernel, all on `stream`.
// ---------------------------------------------------------------------------
extern "C" void kernel_launch(void* const* d_in, const int* in_sizes, int n_in,
                              void* d_out, int out_size, void* d_ws, size_t ws_size,
                              hipStream_t stream) {
  _Float16* w = (_Float16*)d_ws;   // needs W_TOTAL*2 = ~2.1 MB of scratch

  auto cvt = [&](int idx, size_t off, int rs, int cs, int rd, int cd, int srcElemOff = 0) {
    const int n = rd * cd;
    cvt_pad_f16<<<(n + 255) / 256, 256, 0, stream>>>(
        (const float*)d_in[idx] + srcElemOff, w + off, rs, cs, rd, cd);
  };
  cvt(1,  OFF_ENC1, 128, 64, 128, 64);
  cvt(3,  OFF_ENC2, 256, 128, 256, 128);
  cvt(5,  OFF_FC1,  128, 256, 128, 256);
  cvt(7,  OFF_FC2,  128, 128, 128, 128);
  cvt(9,  OFF_FC3,  256, 128, 256, 128);
  cvt(11, OFF_POL1, 128, 4, 128, 32);          // pad K 4 -> 32
  cvt(13, OFF_POL2, 128, 128, 128, 128);
  cvt(15, OFF_GATE, 7, 384, 16, 384);          // pad N 7 -> 16
  cvt(17, OFF_E1, 7 * 256, 256, 7 * 256, 256);
  cvt(19, OFF_E2, 7 * 256, 256, 7 * 256, 256);
  for (int e = 0; e < 7; ++e)                  // pad N 1 -> 16 per expert
    cvt(21, OFF_FIN + (size_t)e * 16 * 256, 1, 256, 16, 256, e * 256);

  MoEParams p{};
  p.x      = (const float*)d_in[0];
  p.b_enc1 = (const float*)d_in[2];
  p.b_enc2 = (const float*)d_in[4];
  p.b_fc1  = (const float*)d_in[6];
  p.b_fc2  = (const float*)d_in[8];
  p.b_fc3  = (const float*)d_in[10];
  p.b_pol1 = (const float*)d_in[12];
  p.b_pol2 = (const float*)d_in[14];
  p.b_gate = (const float*)d_in[16];
  p.b_e1   = (const float*)d_in[18];
  p.b_e2   = (const float*)d_in[20];
  p.b_fin  = (const float*)d_in[22];
  p.w      = w;
  p.out    = (float*)d_out;

  const int Btot    = in_sizes[0] / 4;   // 65536
  const int nBlocks = Btot / BROWS;      // 512

  (void)hipFuncSetAttribute((const void*)moe_inr_fused,
                            hipFuncAttributeMaxDynamicSharedMemorySize,
                            (int)SM_TOTAL);
  moe_inr_fused<<<nBlocks, 256, SM_TOTAL, stream>>>(p);
}
